// JSA_transformer_72499047956648
// MI455X (gfx1250) — matrix operand order repaired and checked by reading, not compiled, not run on hardware
//
#include <hip/hip_runtime.h>
#include <hip/hip_bf16.h>
#include <math.h>

#define RESV   512
#define DIMC   64
#define NTOK   64          // tokens per 8x8 window
#define NWIN   4096
#define LTOK   (RESV*RESV) // 262144
#define HIDC   256
#define LNEPS  1e-5f
#define QSCALE 0.17677669529663687f  // 1/sqrt(32)

typedef __attribute__((ext_vector_type(16))) __bf16 v16bf;
typedef __attribute__((ext_vector_type(8)))  __bf16 v8bf;
typedef __attribute__((ext_vector_type(8)))  float  v8f;
typedef __attribute__((ext_vector_type(4)))  float  v4f;

// ---------- scalar bf16 <-> f32 (round-to-nearest-even-ish) ----------
__device__ inline __bf16 f2bf(float f) {
    unsigned u = __builtin_bit_cast(unsigned, f);
    unsigned r = u + 0x7fffu + ((u >> 16) & 1u);
    unsigned short h = (unsigned short)(r >> 16);
    return __builtin_bit_cast(__bf16, h);
}
__device__ inline float bf2f(__bf16 b) {
    unsigned short h = __builtin_bit_cast(unsigned short, b);
    unsigned u = ((unsigned)h) << 16;
    return __builtin_bit_cast(float, u);
}
__device__ inline v8f vzero8() {
    v8f z;
#pragma unroll
    for (int i = 0; i < 8; ++i) z[i] = 0.f;
    return z;
}

// ---------- WMMA wrapper: D = A(16x32 bf16) x B(32x16 bf16) + C(f32) ----------
__device__ inline v8f wmma_bf16(v16bf a, v16bf b, v8f c) {
    return __builtin_amdgcn_wmma_f32_16x16x32_bf16(false, a, false, b, (short)0, c,
                                                   false, false);
}

// A fragment (16x32, row-major [M][K] source in LDS, stride in halves).
// lane: M = lane&15, hsel = lane>>4; elem i: K=kcol+hsel*16+i
__device__ inline v16bf lds_load_A(const __bf16* base, int stride, int mrow, int kcol, int lane) {
    int M = lane & 15, hs = lane >> 4;
    const __bf16* p = base + (size_t)(mrow + M) * stride + kcol + hs * 16;
    v8bf lo = *(const v8bf*)(p);
    v8bf hi = *(const v8bf*)(p + 8);
    v16bf a;
#pragma unroll
    for (int i = 0; i < 8; ++i) { a[i] = lo[i]; a[i + 8] = hi[i]; }
    return a;
}

// B fragment (32x16) where the LDS source is stored [N][K] row-major (i.e. transposed
// operand: K^T from row-major K, or V from V^T). lane: N = nrow + (lane&15);
// elem i: K = kcol + (lane>>4)*16 + i  (contiguous along the source row).
__device__ inline v16bf lds_load_Bt(const __bf16* base, int stride, int nrow, int kcol, int lane) {
    int N = lane & 15, hs = lane >> 4;
    const __bf16* p = base + (size_t)(nrow + N) * stride + kcol + hs * 16;
    v8bf lo = *(const v8bf*)(p);
    v8bf hi = *(const v8bf*)(p + 8);
    v16bf b;
#pragma unroll
    for (int i = 0; i < 8; ++i) { b[i] = lo[i]; b[i + 8] = hi[i]; }
    return b;
}

// B fragment from pre-packed global weights: one contiguous 32B load per lane.
__device__ inline v16bf ldg_B(const __bf16* pw, int kt, int nt, int ntiles, int lane) {
    const __bf16* p = pw + ((size_t)(kt * ntiles + nt) * 32 + lane) * 16;
    v8bf lo = *(const v8bf*)(p);
    v8bf hi = *(const v8bf*)(p + 8);
    v16bf b;
#pragma unroll
    for (int i = 0; i < 8; ++i) { b[i] = lo[i]; b[i + 8] = hi[i]; }
    return b;
}

// window token -> global token (window w = wy*64+wx, token n = iy*8+ix)
__device__ inline int win_tok(int w, int n) {
    return (((w >> 6) * 8 + (n >> 3)) << 9) + ((w & 63) * 8 + (n & 7));
}

// ---------- weight pre-pack: f32 [K][N] -> bf16 B-fragment order ----------
__global__ void pack_w_kernel(const float* __restrict__ src, __bf16* __restrict__ dst,
                              int K, int N) {
    int gid = blockIdx.x * blockDim.x + threadIdx.x;
    if (gid >= K * N) return;
    int i    = gid & 15;
    int lane = (gid >> 4) & 31;
    int rest = gid >> 9;
    int NT   = N >> 4;
    int nt   = rest % NT;
    int kt   = rest / NT;
    int k = kt * 32 + (lane >> 4) * 16 + i;
    int n = nt * 16 + (lane & 15);
    dst[gid] = f2bf(src[(size_t)k * N + n]);
}

// ---------- relative position bias table: [2][64][64] ----------
__global__ void bias_tab_kernel(const float* __restrict__ rpb, float* __restrict__ bt) {
    int gid = blockIdx.x * blockDim.x + threadIdx.x;
    if (gid >= 2 * 64 * 64) return;
    int h = gid >> 12;
    int n = (gid >> 6) & 63;
    int m = gid & 63;
    int dy = (n >> 3) - (m >> 3) + 7;
    int dx = (n & 7) - (m & 7) + 7;
    bt[gid] = rpb[(dy * 15 + dx) * 2 + h];
}

// ---------- fused window attention ----------
__global__ __launch_bounds__(128) void attn_kernel(
    const float* __restrict__ x,   const float* __restrict__ f,
    const float* __restrict__ n1w, const float* __restrict__ n1b,
    const float* __restrict__ nfw, const float* __restrict__ nfb,
    const __bf16* __restrict__ pwq,   const float* __restrict__ bq,
    const __bf16* __restrict__ pwkv,  const float* __restrict__ bkv,
    const __bf16* __restrict__ pwqf,  const float* __restrict__ bqf,
    const __bf16* __restrict__ pwkvf, const float* __restrict__ bkvf,
    const __bf16* __restrict__ pwproj,const float* __restrict__ bproj,
    const float* __restrict__ biasTab,
    float* __restrict__ x2) {
    __shared__ __bf16 sX [NTOK][72];
    __shared__ __bf16 sF [NTOK][72];
    __shared__ __bf16 sQ [NTOK][72];   // reused as softmax P (head 0)
    __shared__ __bf16 sK [NTOK][72];
    __shared__ __bf16 sVt[DIMC][72];   // V transposed: [d][token]
    __shared__ __bf16 sQf[NTOK][72];   // reused as softmax P (head 1)
    __shared__ __bf16 sKf[NTOK][72];
    __shared__ __bf16 sO [NTOK][72];
    __shared__ float  sS [2][NTOK][68];

    const int tid  = threadIdx.x;
    const int lane = tid & 31;
    const int wave = tid >> 5;
    const int w    = blockIdx.x;

    // ---- LayerNorm(x)/LayerNorm(f), window-gathered into LDS bf16 ----
    {
        int r = tid & 63;
        bool isF = tid >= 64;
        const float* src = isF ? f : x;
        const float* gw  = isF ? nfw : n1w;
        const float* gb  = isF ? nfb : n1b;
        const float* row = src + (size_t)win_tok(w, r) * DIMC;
        float vals[64];
        float mean = 0.f;
#pragma unroll
        for (int i = 0; i < 16; ++i) {
            v4f q = *(const v4f*)(row + i * 4);
#pragma unroll
            for (int j = 0; j < 4; ++j) { vals[i * 4 + j] = q[j]; mean += q[j]; }
        }
        mean *= (1.f / 64.f);
        float var = 0.f;
#pragma unroll
        for (int c = 0; c < 64; ++c) { float d = vals[c] - mean; var += d * d; }
        var *= (1.f / 64.f);
        float inv = rsqrtf(var + LNEPS);
        __bf16* drow = isF ? sF[r] : sX[r];
#pragma unroll
        for (int c = 0; c < 64; ++c)
            drow[c] = f2bf((vals[c] - mean) * inv * gw[c] + gb[c]);
    }
    __syncthreads();

    // ---- QKV projections: Q,K,V (from x) and Qf,Kf (from f); 80 16x16 tiles ----
    for (int j = wave; j < 80; j += 4) {
        int mat  = j / 16;       // 0:Q 1:K 2:V 3:Qf 4:Kf
        int tile = j & 15;
        int mt = tile >> 2, nt = tile & 3;
        __bf16 (*in)[72] = (mat >= 3) ? sF : sX;
        const __bf16* pw; const float* bias; int ntiles, wnt; float scale;
        switch (mat) {
            case 0:  pw = pwq;   bias = bq   + nt * 16;      ntiles = 4; wnt = nt;     scale = QSCALE; break;
            case 1:  pw = pwkv;  bias = bkv  + nt * 16;      ntiles = 8; wnt = nt;     scale = 1.f;    break;
            case 2:  pw = pwkv;  bias = bkv  + 64 + nt * 16; ntiles = 8; wnt = nt + 4; scale = 1.f;    break;
            case 3:  pw = pwqf;  bias = bqf  + nt * 16;      ntiles = 4; wnt = nt;     scale = QSCALE; break;
            default: pw = pwkvf; bias = bkvf + nt * 16;      ntiles = 8; wnt = nt;     scale = 1.f;    break;
        }
        v8f acc = vzero8();
#pragma unroll
        for (int kt = 0; kt < 2; ++kt) {
            v16bf a = lds_load_A(&in[0][0], 72, mt * 16, kt * 32, lane);
            v16bf b = ldg_B(pw, kt, wnt, ntiles, lane);
            acc = wmma_bf16(a, b, acc);
        }
        float bn = bias[lane & 15];
        int hs = lane >> 4;
        int ncol = nt * 16 + (lane & 15);
        if (mat == 2) {  // store V transposed: contiguous v8bf row store
            v8bf vv;
#pragma unroll
            for (int v = 0; v < 8; ++v) vv[v] = f2bf(acc[v] + bn);
            *(v8bf*)&sVt[ncol][mt * 16 + 8 * hs] = vv;
        } else {
            __bf16 (*dst)[72] = (mat == 0) ? sQ : (mat == 1) ? sK : (mat == 3) ? sQf : sKf;
#pragma unroll
            for (int v = 0; v < 8; ++v)
                dst[mt * 16 + 8 * hs + v][ncol] = f2bf((acc[v] + bn) * scale);
        }
    }
    __syncthreads();

    // ---- joint scores: S = (Q K^T + bias) * (Qf Kf^T + bias), K-dim = head dim = 32 ----
    for (int j = wave; j < 32; j += 4) {
        int h = j >> 4;
        int tile = j & 15;
        int mt = tile >> 2, nt = tile & 3;
        v16bf aq  = lds_load_A (&sQ [0][0], 72, mt * 16, h * 32, lane);
        v16bf bk  = lds_load_Bt(&sK [0][0], 72, nt * 16, h * 32, lane);
        v8f s1 = wmma_bf16(aq, bk, vzero8());
        v16bf aqf = lds_load_A (&sQf[0][0], 72, mt * 16, h * 32, lane);
        v16bf bkf = lds_load_Bt(&sKf[0][0], 72, nt * 16, h * 32, lane);
        v8f s2 = wmma_bf16(aqf, bkf, vzero8());
        int hs = lane >> 4;
        int ncol = nt * 16 + (lane & 15);
        const float* bt = biasTab + ((size_t)h * NTOK + mt * 16 + 8 * hs) * NTOK + ncol;
#pragma unroll
        for (int v = 0; v < 8; ++v) {
            float b = bt[(size_t)v * NTOK];
            sS[h][mt * 16 + 8 * hs + v][ncol] = (s1[v] + b) * (s2[v] + b);
        }
    }
    __syncthreads();

    // ---- softmax rows (128 threads = 2 heads x 64 rows), P -> bf16 into sQ/sQf ----
    {
        int h = tid >> 6;
        int rr = tid & 63;
        float* srow = sS[h][rr];
        float mx = -1e30f;
#pragma unroll
        for (int i = 0; i < 64; ++i) mx = fmaxf(mx, srow[i]);
        float e[64];
        float sum = 0.f;
#pragma unroll
        for (int i = 0; i < 64; ++i) { e[i] = __expf(srow[i] - mx); sum += e[i]; }
        float inv = 1.f / sum;
        __bf16* prow = (h == 0) ? sQ[rr] : sQf[rr];
#pragma unroll
        for (int i = 0; i < 64; ++i) prow[i] = f2bf(e[i] * inv);
    }
    __syncthreads();

    // ---- O = P @ V (per head, N = 32) ----
    for (int j = wave; j < 16; j += 4) {
        int h = j >> 3;
        int tile = j & 7;
        int mt = tile >> 1, nt = tile & 1;
        __bf16 (*P)[72] = h ? sQf : sQ;
        v8f acc = vzero8();
#pragma unroll
        for (int kt = 0; kt < 2; ++kt) {
            v16bf a = lds_load_A (&P[0][0],   72, mt * 16, kt * 32, lane);
            v16bf b = lds_load_Bt(&sVt[0][0], 72, h * 32 + nt * 16, kt * 32, lane);
            acc = wmma_bf16(a, b, acc);
        }
        int hs = lane >> 4;
        int ncol = h * 32 + nt * 16 + (lane & 15);
#pragma unroll
        for (int v = 0; v < 8; ++v)
            sO[mt * 16 + 8 * hs + v][ncol] = f2bf(acc[v]);
    }
    __syncthreads();

    // ---- proj + shortcut residual, scatter back to token order ----
    for (int j = wave; j < 16; j += 4) {
        int mt = j >> 2, nt = j & 3;
        v8f acc = vzero8();
#pragma unroll
        for (int kt = 0; kt < 2; ++kt) {
            v16bf a = lds_load_A(&sO[0][0], 72, mt * 16, kt * 32, lane);
            v16bf b = ldg_B(pwproj, kt, nt, 4, lane);
            acc = wmma_bf16(a, b, acc);
        }
        int hs = lane >> 4;
        int ncol = nt * 16 + (lane & 15);
        float bn = bproj[ncol];
#pragma unroll
        for (int v = 0; v < 8; ++v) {
            int t = win_tok(w, mt * 16 + 8 * hs + v);
            size_t idx = (size_t)t * DIMC + ncol;
            x2[idx] = acc[v] + bn + x[idx];
        }
    }
}

// ---------- LayerNorm2: x2 f32 -> bf16 ----------
__global__ void ln2_kernel(const float* __restrict__ x2, const float* __restrict__ gw,
                           const float* __restrict__ gb, __bf16* __restrict__ out) {
    int t = blockIdx.x * blockDim.x + threadIdx.x;
    if (t >= LTOK) return;
    const float* row = x2 + (size_t)t * DIMC;
    float vals[64];
    float mean = 0.f;
#pragma unroll
    for (int i = 0; i < 16; ++i) {
        v4f q = *(const v4f*)(row + i * 4);
#pragma unroll
        for (int j = 0; j < 4; ++j) { vals[i * 4 + j] = q[j]; mean += q[j]; }
    }
    mean *= (1.f / 64.f);
    float var = 0.f;
#pragma unroll
    for (int c = 0; c < 64; ++c) { float d = vals[c] - mean; var += d * d; }
    var *= (1.f / 64.f);
    float inv = rsqrtf(var + LNEPS);
    __bf16* drow = out + (size_t)t * DIMC;
#pragma unroll
    for (int c = 0; c < 64; ++c)
        drow[c] = f2bf((vals[c] - mean) * inv * gw[c] + gb[c]);
}

// ---------- fc1 GEMM: [L,64] bf16 x [64,256] -> bf16, LDS-staged coalesced I/O ----------
__global__ __launch_bounds__(128) void fc1_kernel(const __bf16* __restrict__ a,
                                                  const __bf16* __restrict__ pw,
                                                  const float* __restrict__ bias,
                                                  __bf16* __restrict__ out) {
    __shared__ __bf16 sA[64][72];
    __shared__ __bf16 sOut[64][64];
    const int tid = threadIdx.x, lane = tid & 31, wave = tid >> 5;
    const size_t m0 = (size_t)blockIdx.x * 64;
    const int nblk = blockIdx.y;  // 0..3, 64 output cols each
    {
        int r = tid >> 1, half = (tid & 1) * 32;
        const __bf16* src = a + (m0 + r) * 64 + half;
#pragma unroll
        for (int i = 0; i < 4; ++i)
            *(v8bf*)&sA[r][half + i * 8] = *(const v8bf*)(src + i * 8);
    }
    __syncthreads();
    for (int tile = wave; tile < 16; tile += 4) {
        int mt = tile >> 2, nt = tile & 3;
        v8f acc = vzero8();
#pragma unroll
        for (int kt = 0; kt < 2; ++kt) {
            v16bf av = lds_load_A(&sA[0][0], 72, mt * 16, kt * 32, lane);
            v16bf bv = ldg_B(pw, kt, nblk * 4 + nt, 16, lane);
            acc = wmma_bf16(av, bv, acc);
        }
        int hs = lane >> 4, ncol = nt * 16 + (lane & 15);
        float bn = bias[nblk * 64 + ncol];
#pragma unroll
        for (int v = 0; v < 8; ++v)
            sOut[mt * 16 + 8 * hs + v][ncol] = f2bf(acc[v] + bn);
    }
    __syncthreads();
    {
        int r = tid >> 1, half = (tid & 1) * 32;
        __bf16* dst = out + (m0 + r) * HIDC + nblk * 64 + half;
#pragma unroll
        for (int i = 0; i < 4; ++i)
            *(v8bf*)(dst + i * 8) = *(const v8bf*)&sOut[r][half + i * 8];
    }
}

// ---------- depthwise 3x3 conv + bias + relu (channel-innermost layout) ----------
__global__ void dwconv_kernel(const __bf16* __restrict__ y1, const float* __restrict__ cw,
                              const float* __restrict__ cb, __bf16* __restrict__ y2) {
    int gid = blockIdx.x * blockDim.x + threadIdx.x;  // L * 64 threads (4 ch each)
    int cg = gid & 63;
    int t  = gid >> 6;
    if (t >= LTOK) return;
    int y0 = t >> 9, x0 = t & 511;
    int c0 = cg * 4;
    float acc[4];
#pragma unroll
    for (int i = 0; i < 4; ++i) acc[i] = cb[c0 + i];
#pragma unroll
    for (int ky = 0; ky < 3; ++ky) {
        int yy = y0 + ky - 1;
        if (yy < 0 || yy >= RESV) continue;
#pragma unroll
        for (int kx = 0; kx < 3; ++kx) {
            int xx = x0 + kx - 1;
            if (xx < 0 || xx >= RESV) continue;
            const __bf16* p = y1 + (size_t)(yy * RESV + xx) * HIDC + c0;
#pragma unroll
            for (int i = 0; i < 4; ++i)
                acc[i] += bf2f(p[i]) * cw[(c0 + i) * 9 + ky * 3 + kx];
        }
    }
    __bf16* d = y2 + (size_t)t * HIDC + c0;
#pragma unroll
    for (int i = 0; i < 4; ++i) d[i] = f2bf(fmaxf(acc[i], 0.f));
}

// ---------- fc2 GEMM: [L,256] bf16 x [256,64] + bias + residual -> f32 out ----------
__global__ __launch_bounds__(128) void fc2_kernel(const __bf16* __restrict__ a,
                                                  const __bf16* __restrict__ pw,
                                                  const float* __restrict__ bias,
                                                  const float* __restrict__ x2,
                                                  float* __restrict__ out) {
    __shared__ __bf16 sA[64][264];
    const int tid = threadIdx.x, lane = tid & 31, wave = tid >> 5;
    const size_t m0 = (size_t)blockIdx.x * 64;
    {
        int r = tid >> 1, half = (tid & 1) * 128;
        const __bf16* src = a + (m0 + r) * HIDC + half;
#pragma unroll
        for (int i = 0; i < 16; ++i)
            *(v8bf*)&sA[r][half + i * 8] = *(const v8bf*)(src + i * 8);
    }
    __syncthreads();
    for (int tile = wave; tile < 16; tile += 4) {
        int mt = tile >> 2, nt = tile & 3;
        v8f acc = vzero8();
#pragma unroll
        for (int kt = 0; kt < 8; ++kt) {
            v16bf av = lds_load_A(&sA[0][0], 264, mt * 16, kt * 32, lane);
            v16bf bv = ldg_B(pw, kt, nt, 4, lane);
            acc = wmma_bf16(av, bv, acc);
        }
        int hs = lane >> 4, ncol = nt * 16 + (lane & 15);
        float bn = bias[ncol];
#pragma unroll
        for (int v = 0; v < 8; ++v) {
            size_t idx = (m0 + mt * 16 + 8 * hs + v) * DIMC + ncol;
            out[idx] = acc[v] + bn + x2[idx];
        }
    }
}

extern "C" void kernel_launch(void* const* d_in, const int* in_sizes, int n_in,
                              void* d_out, int out_size, void* d_ws, size_t ws_size,
                              hipStream_t stream) {
    (void)in_sizes; (void)n_in; (void)out_size; (void)ws_size;
    const float* x      = (const float*)d_in[0];
    const float* f      = (const float*)d_in[1];
    const float* n1w    = (const float*)d_in[2];
    const float* n1b    = (const float*)d_in[3];
    const float* nfw    = (const float*)d_in[4];
    const float* nfb    = (const float*)d_in[5];
    const float* wq_x   = (const float*)d_in[6];
    const float* bq_x   = (const float*)d_in[7];
    const float* wkv_x  = (const float*)d_in[8];
    const float* bkv_x  = (const float*)d_in[9];
    const float* wq_f   = (const float*)d_in[10];
    const float* bq_f   = (const float*)d_in[11];
    const float* wkv_f  = (const float*)d_in[12];
    const float* bkv_f  = (const float*)d_in[13];
    const float* rpb    = (const float*)d_in[14];
    const float* proj_w = (const float*)d_in[15];
    const float* proj_b = (const float*)d_in[16];
    const float* n2w    = (const float*)d_in[17];
    const float* n2b    = (const float*)d_in[18];
    const float* fc1_w  = (const float*)d_in[19];
    const float* fc1_b  = (const float*)d_in[20];
    const float* conv_w = (const float*)d_in[21];
    const float* conv_b = (const float*)d_in[22];
    const float* fc2_w  = (const float*)d_in[23];
    const float* fc2_b  = (const float*)d_in[24];

    char* ws = (char*)d_ws;
    size_t off = 0;
    auto take = [&](size_t bytes) -> char* {
        char* p = ws + off;
        off = (off + bytes + 255) & ~(size_t)255;
        return p;
    };
    __bf16* pw_qx   = (__bf16*)take((size_t)64 * 64 * 2);
    __bf16* pw_kvx  = (__bf16*)take((size_t)64 * 128 * 2);
    __bf16* pw_qf   = (__bf16*)take((size_t)64 * 64 * 2);
    __bf16* pw_kvf  = (__bf16*)take((size_t)64 * 128 * 2);
    __bf16* pw_proj = (__bf16*)take((size_t)64 * 64 * 2);
    __bf16* pw_fc1  = (__bf16*)take((size_t)64 * 256 * 2);
    __bf16* pw_fc2  = (__bf16*)take((size_t)256 * 64 * 2);
    float*  biasTab = (float*)take((size_t)2 * 64 * 64 * 4);
    float*  x2      = (float*)take((size_t)LTOK * 64 * 4);
    __bf16* x2n     = (__bf16*)take((size_t)LTOK * 64 * 2);
    __bf16* y1      = (__bf16*)take((size_t)LTOK * 256 * 2);
    __bf16* y2      = (__bf16*)take((size_t)LTOK * 256 * 2);

    pack_w_kernel<<<16, 256, 0, stream>>>(wq_x,   pw_qx,   64, 64);
    pack_w_kernel<<<32, 256, 0, stream>>>(wkv_x,  pw_kvx,  64, 128);
    pack_w_kernel<<<16, 256, 0, stream>>>(wq_f,   pw_qf,   64, 64);
    pack_w_kernel<<<32, 256, 0, stream>>>(wkv_f,  pw_kvf,  64, 128);
    pack_w_kernel<<<16, 256, 0, stream>>>(proj_w, pw_proj, 64, 64);
    pack_w_kernel<<<64, 256, 0, stream>>>(fc1_w,  pw_fc1,  64, 256);
    pack_w_kernel<<<64, 256, 0, stream>>>(fc2_w,  pw_fc2,  256, 64);
    bias_tab_kernel<<<32, 256, 0, stream>>>(rpb, biasTab);

    attn_kernel<<<NWIN, 128, 0, stream>>>(x, f, n1w, n1b, nfw, nfb,
                                          pw_qx, bq_x, pw_kvx, bkv_x,
                                          pw_qf, bq_f, pw_kvf, bkv_f,
                                          pw_proj, proj_b, biasTab, x2);

    ln2_kernel<<<LTOK / 256, 256, 0, stream>>>(x2, n2w, n2b, x2n);
    fc1_kernel<<<dim3(LTOK / 64, 4), 128, 0, stream>>>(x2n, pw_fc1, fc1_b, y1);
    dwconv_kernel<<<(LTOK * 64) / 256, 256, 0, stream>>>(y1, conv_w, conv_b, y2);
    fc2_kernel<<<LTOK / 64, 128, 0, stream>>>(y2, pw_fc2, fc2_b, x2, (float*)d_out);
}
